// Ablation_StandardCNN_MoE_20718922236399
// MI455X (gfx1250) — matrix operand order repaired and checked by reading, not compiled, not run on hardware
//
#include <hip/hip_runtime.h>
#include <math.h>

// ---------------------------------------------------------------------------
// Model constants (from reference)
// ---------------------------------------------------------------------------
#define BB    128
#define NB    55
#define NC    8
#define TT    512
#define DD    128
#define SS    64
#define LL    4
#define HEADS 8
#define DH    16
#define EE    8
#define KTOP  2
#define HID   512
#define NTOK  (BB * SS)        // 8192
#define CIN   (NB * NC)        // 440
#define KCONV (CIN * 32)       // 14080  (kernel padded 31 -> 32)

typedef __attribute__((ext_vector_type(16))) __bf16 v16bf;
typedef __attribute__((ext_vector_type(8)))  __bf16 v8bf;
typedef __attribute__((ext_vector_type(8)))  float  v8f;

__device__ __forceinline__ float gelu_exact(float x) {
    return 0.5f * x * (1.0f + erff(x * 0.70710678118654752f));
}

// ---------------------------------------------------------------------------
// Fragment load + WMMA macro (shared by generic GEMM and conv GEMM).
// LDS layouts: As[row][k] row-major (64x32), Bt[n][k] (B transposed, 64x32).
// A frag (16x32 bf16): lanes 0-15 -> M=lane, K chunks {0..7},{16..23};
//                      lanes16-31 -> M=lane-16, K chunks {8..15},{24..31}.
// B frag (32x16):      lanes 0-15 -> N=lane, K chunks {0..7},{16..23};
//                      lanes16-31 -> N=lane-16, K chunks {8..15},{24..31}.
// ---------------------------------------------------------------------------
#define GEMM_FRAGS_AND_MMA()                                                   \
    do {                                                                       \
        v16bf afrag[2], bfrag[2];                                              \
        _Pragma("unroll")                                                      \
        for (int fi = 0; fi < 2; ++fi) {                                       \
            int arow = wm * 32 + fi * 16 + (lane & 15);                        \
            int ako  = (lane < 16) ? 0 : 8;                                    \
            v8bf lo = *(const v8bf*)&As[arow * 32 + ako];                      \
            v8bf hi = *(const v8bf*)&As[arow * 32 + ako + 16];                 \
            _Pragma("unroll")                                                  \
            for (int j = 0; j < 8; ++j) {                                      \
                afrag[fi][j] = lo[j]; afrag[fi][8 + j] = hi[j];                \
            }                                                                  \
        }                                                                      \
        _Pragma("unroll")                                                      \
        for (int fi = 0; fi < 2; ++fi) {                                       \
            int bcol = wn * 32 + fi * 16 + (lane & 15);                        \
            int bko  = (lane < 16) ? 0 : 8;                                    \
            v8bf lo = *(const v8bf*)&Bt[bcol * 32 + bko];                      \
            v8bf hi = *(const v8bf*)&Bt[bcol * 32 + bko + 16];                 \
            _Pragma("unroll")                                                  \
            for (int j = 0; j < 8; ++j) {                                      \
                bfrag[fi][j] = lo[j]; bfrag[fi][8 + j] = hi[j];                \
            }                                                                  \
        }                                                                      \
        _Pragma("unroll")                                                      \
        for (int fi = 0; fi < 2; ++fi)                                         \
            _Pragma("unroll")                                                  \
            for (int fj = 0; fj < 2; ++fj)                                     \
                acc[fi][fj] = __builtin_amdgcn_wmma_f32_16x16x32_bf16(         \
                    false, afrag[fi], false, bfrag[fj], (short)0,              \
                    acc[fi][fj], false, false);                                \
    } while (0)

// ---------------------------------------------------------------------------
// Generic bf16 GEMM: C(MxN) = A(MxK,row) * B(KxN,row) with epilogue variants.
// Block: 128 threads (4 waves), 64x64 tile; per-wave 32x32 (2x2 WMMA frags).
// EPI 0: Cf = acc + bias            (QKV projections)
// EPI 1: Cf += acc + bias           (attention out projection residual)
// EPI 2: Cb = bf16(gelu(acc+bias))  (MoE up-proj)
// EPI 3: Cf += cw[row]*(acc+bias)   (MoE down-proj, fused top-k combine)
// M, N multiples of 64; K multiple of 32.
// ---------------------------------------------------------------------------
template <int EPI>
__global__ void __launch_bounds__(128)
k_gemm(const __bf16* __restrict__ A, const __bf16* __restrict__ Bm,
       const float* __restrict__ bias, float* __restrict__ Cf,
       __bf16* __restrict__ Cb, const float* __restrict__ cw, int cwStride,
       int M, int N, int K) {
    __shared__ __bf16 As[64 * 32];
    __shared__ __bf16 Bt[64 * 32];
    const int tid  = threadIdx.x;
    const int lane = tid & 31, wave = tid >> 5;
    const int wm = wave >> 1, wn = wave & 1;
    const int m0 = blockIdx.y * 64, n0 = blockIdx.x * 64;

    v8f acc[2][2] = {};

    for (int k0 = 0; k0 < K; k0 += 32) {
        // ---- stage A tile 64x32 (16B vector loads) ----
        for (int i = tid; i < 256; i += 128) {
            int row = i >> 2, cc = i & 3;
            *(uint4*)&As[row * 32 + cc * 8] =
                *(const uint4*)&A[(size_t)(m0 + row) * K + k0 + cc * 8];
        }
        // ---- stage B tile 32x64, transposed on store -> Bt[n][k] ----
        for (int i = tid; i < 256; i += 128) {
            int kk = i >> 3, nc = i & 7;
            v8bf bv = *(const v8bf*)&Bm[(size_t)(k0 + kk) * N + n0 + nc * 8];
#pragma unroll
            for (int j = 0; j < 8; ++j) Bt[(nc * 8 + j) * 32 + kk] = bv[j];
        }
        __syncthreads();

        GEMM_FRAGS_AND_MMA();

        if (k0 + 32 < K) {  // gfx1250 global_prefetch_b8 of next A tile
            __builtin_prefetch(&A[(size_t)(m0 + (tid >> 1)) * K + k0 + 32 +
                                  (tid & 1) * 16], 0, 0);
        }
        __syncthreads();
    }

    // ---- epilogue: C 16x16 f32 layout: VGPR r -> (M=r | r+8, N=lane&15) ----
#pragma unroll
    for (int i = 0; i < 2; ++i)
#pragma unroll
        for (int j = 0; j < 2; ++j)
#pragma unroll
            for (int r = 0; r < 8; ++r) {
                int row = m0 + wm * 32 + i * 16 + r + ((lane & 16) ? 8 : 0);
                int col = n0 + wn * 32 + j * 16 + (lane & 15);
                float v = acc[i][j][r] + bias[col];
                if (EPI == 0)      Cf[(size_t)row * N + col] = v;
                else if (EPI == 1) Cf[(size_t)row * N + col] += v;
                else if (EPI == 2) Cb[(size_t)row * N + col] = (__bf16)gelu_exact(v);
                else               Cf[(size_t)row * N + col] += cw[(size_t)row * cwStride] * v;
            }
}

// ---------------------------------------------------------------------------
// Conv1d (440ch, k=31, stride 8, pad 15) as implicit GEMM via WMMA.
// K index = c*32 + kk (kernel padded to 32; tap 31 has zero weight).
// One block row = one batch element (S=64 rows). A gathered with time padding.
// ---------------------------------------------------------------------------
__global__ void __launch_bounds__(128)
k_conv_gemm(const __bf16* __restrict__ xb, const __bf16* __restrict__ wb,
            float* __restrict__ out) {
    __shared__ __bf16 As[64 * 32];
    __shared__ __bf16 Bt[64 * 32];
    const int tid  = threadIdx.x;
    const int lane = tid & 31, wave = tid >> 5;
    const int wm = wave >> 1, wn = wave & 1;
    const int b  = blockIdx.y;
    const int n0 = blockIdx.x * 64;

    v8f acc[2][2] = {};

    for (int c = 0; c < CIN; ++c) {
        // A: rows s=0..63, cols kk=0..31, value = xb[b][c][s*8-15+kk] (0-pad)
        {
            int s  = tid >> 1;
            int kb = (tid & 1) * 16;
            const __bf16* src = xb + ((size_t)b * CIN + c) * TT;
#pragma unroll
            for (int u = 0; u < 16; ++u) {
                int kk = kb + u;
                int t  = s * 8 - 15 + kk;
                As[s * 32 + kk] = (t >= 0 && t < TT) ? src[t] : (__bf16)0.0f;
            }
        }
        // B: wb layout [c][kk][128] (row-major KxN) -> transpose into Bt
        for (int i = tid; i < 256; i += 128) {
            int kk = i >> 3, nc = i & 7;
            v8bf bv = *(const v8bf*)&wb[((size_t)c * 32 + kk) * DD + n0 + nc * 8];
#pragma unroll
            for (int j = 0; j < 8; ++j) Bt[(nc * 8 + j) * 32 + kk] = bv[j];
        }
        __syncthreads();

        GEMM_FRAGS_AND_MMA();

        __syncthreads();
    }

#pragma unroll
    for (int i = 0; i < 2; ++i)
#pragma unroll
        for (int j = 0; j < 2; ++j)
#pragma unroll
            for (int r = 0; r < 8; ++r) {
                int row = wm * 32 + i * 16 + r + ((lane & 16) ? 8 : 0);
                int col = n0 + wn * 32 + j * 16 + (lane & 15);
                out[((size_t)b * SS + row) * DD + col] = acc[i][j][r];
            }
}

// ---------------------------------------------------------------------------
// Prep / elementwise kernels
// ---------------------------------------------------------------------------
__global__ void k_prep_x(const float* __restrict__ x, __bf16* __restrict__ xb) {
    size_t idx = (size_t)blockIdx.x * blockDim.x + threadIdx.x;
    if (idx >= (size_t)BB * CIN * TT) return;
    int t = idx % TT;
    int c = (idx / TT) % CIN;
    int b = idx / ((size_t)TT * CIN);
    int nb = c >> 3, nc = c & 7;
    // x is [B, NB, T, NC]
    xb[idx] = (__bf16)x[(((size_t)b * NB + nb) * TT + t) * NC + nc];
}

__global__ void k_prep_convw(const float* __restrict__ w, __bf16* __restrict__ wb) {
    size_t idx = (size_t)blockIdx.x * blockDim.x + threadIdx.x;
    if (idx >= (size_t)CIN * 32 * DD) return;
    int d  = idx % DD;
    int kk = (idx / DD) % 32;
    int c  = idx / ((size_t)DD * 32);
    // conv_w is [D, CIN, 31]; pad tap 31 with zero
    wb[idx] = (kk < 31) ? (__bf16)w[((size_t)d * CIN + c) * 31 + kk] : (__bf16)0.0f;
}

__global__ void k_cvt_f32_bf16(const float* __restrict__ s, __bf16* __restrict__ d, int n) {
    int i = blockIdx.x * blockDim.x + threadIdx.x;
    if (i < n) d[i] = (__bf16)s[i];
}

__global__ void k_zero(float* __restrict__ p, int n) {
    int i = blockIdx.x * blockDim.x + threadIdx.x;
    if (i < n) p[i] = 0.0f;
}

// Per-channel batch stats over B*S rows; stat[d]=mean, stat[128+d]=rstd
__global__ void __launch_bounds__(256)
k_bn_stats(const float* __restrict__ hc, float* __restrict__ stat) {
    __shared__ float s1[256], s2[256];
    int d = blockIdx.x, t = threadIdx.x;
    float a = 0.f, b = 0.f;
    for (int m = t; m < NTOK; m += 256) {
        float x = hc[(size_t)m * DD + d];
        a += x; b += x * x;
    }
    s1[t] = a; s2[t] = b;
    __syncthreads();
    for (int st = 128; st > 0; st >>= 1) {
        if (t < st) { s1[t] += s1[t + st]; s2[t] += s2[t + st]; }
        __syncthreads();
    }
    if (t == 0) {
        float mean = s1[0] * (1.0f / NTOK);
        float var  = s2[0] * (1.0f / NTOK) - mean * mean;
        stat[d]       = mean;
        stat[DD + d]  = rsqrtf(var + 1e-5f);
    }
}

// h = gelu(bn(hconv)) + pos   (h laid out [B*S, D])
__global__ void k_bn_act(const float* __restrict__ hc, const float* __restrict__ stat,
                         const float* __restrict__ g, const float* __restrict__ bb,
                         const float* __restrict__ pos, float* __restrict__ h) {
    int idx = blockIdx.x * blockDim.x + threadIdx.x;
    if (idx >= NTOK * DD) return;
    int d = idx & (DD - 1);
    int m = idx >> 7;
    int s = m & (SS - 1);
    float v = (hc[idx] - stat[d]) * stat[DD + d] * g[d] + bb[d];
    h[idx]  = gelu_exact(v) + pos[d * SS + s];  // pos is [1, D, S]
}

// Row LayerNorm -> bf16 tokens
__global__ void k_ln(const float* __restrict__ h, const float* __restrict__ g,
                     const float* __restrict__ bta, __bf16* __restrict__ y) {
    int n = blockIdx.x * blockDim.x + threadIdx.x;
    if (n >= NTOK) return;
    const float* r = h + (size_t)n * DD;
    float m = 0.f;
    for (int d = 0; d < DD; ++d) m += r[d];
    m *= (1.0f / DD);
    float v = 0.f;
    for (int d = 0; d < DD; ++d) { float t = r[d] - m; v += t * t; }
    v *= (1.0f / DD);
    float rs = rsqrtf(v + 1e-5f);
    __bf16* yr = y + (size_t)n * DD;
    for (int d = 0; d < DD; ++d)
        yr[d] = (__bf16)((r[d] - m) * rs * g[d] + bta[d]);
}

// Per (batch, head) attention, one thread per query row (dh=16 is tiny)
__global__ void __launch_bounds__(64)
k_attn(const float* __restrict__ q, const float* __restrict__ k,
       const float* __restrict__ v, __bf16* __restrict__ o) {
    int bh = blockIdx.x;
    int b  = bh >> 3, hh = bh & 7;
    int qi = threadIdx.x;
    const float* qp = q + ((size_t)b * SS + qi) * DD + hh * DH;
    float qr[DH];
#pragma unroll
    for (int d = 0; d < DH; ++d) qr[d] = qp[d];
    float sc[SS];
    float mx = -3.0e38f;
    for (int ki = 0; ki < SS; ++ki) {
        const float* kp = k + ((size_t)b * SS + ki) * DD + hh * DH;
        float s = 0.f;
#pragma unroll
        for (int d = 0; d < DH; ++d) s += qr[d] * kp[d];
        s *= 0.25f;  // 1/sqrt(16)
        sc[ki] = s;
        mx = fmaxf(mx, s);
    }
    float sum = 0.f;
    for (int ki = 0; ki < SS; ++ki) { sc[ki] = __expf(sc[ki] - mx); sum += sc[ki]; }
    float inv = 1.0f / sum;
    float outr[DH] = {};
    for (int ki = 0; ki < SS; ++ki) {
        float p = sc[ki] * inv;
        const float* vp = v + ((size_t)b * SS + ki) * DD + hh * DH;
#pragma unroll
        for (int d = 0; d < DH; ++d) outr[d] += p * vp[d];
    }
    __bf16* op = o + ((size_t)b * SS + qi) * DD + hh * DH;
#pragma unroll
    for (int d = 0; d < DH; ++d) op[d] = (__bf16)outr[d];
}

// Top-2 gating: combine weights + aux-loss accumulators
__global__ void k_gate(const __bf16* __restrict__ y, const float* __restrict__ gw,
                       const float* __restrict__ gb, float* __restrict__ comb,
                       float* __restrict__ auxP, float* __restrict__ auxC) {
    int n = blockIdx.x * blockDim.x + threadIdx.x;
    if (n >= NTOK) return;
    float lg[EE];
#pragma unroll
    for (int e = 0; e < EE; ++e) lg[e] = gb[e];
    const __bf16* tr = y + (size_t)n * DD;
    for (int d = 0; d < DD; ++d) {
        float t = (float)tr[d];
#pragma unroll
        for (int e = 0; e < EE; ++e) lg[e] += t * gw[d * EE + e];
    }
    float mx = lg[0];
#pragma unroll
    for (int e = 1; e < EE; ++e) mx = fmaxf(mx, lg[e]);
    float p[EE], sum = 0.f;
#pragma unroll
    for (int e = 0; e < EE; ++e) { p[e] = __expf(lg[e] - mx); sum += p[e]; }
    float inv = 1.0f / sum;
#pragma unroll
    for (int e = 0; e < EE; ++e) p[e] *= inv;
    int i1 = 0;
#pragma unroll
    for (int e = 1; e < EE; ++e) if (p[e] > p[i1]) i1 = e;
    int i2 = (i1 == 0) ? 1 : 0;
#pragma unroll
    for (int e = 0; e < EE; ++e) if (e != i1 && p[e] > p[i2]) i2 = e;
    float ws = 1.0f / (p[i1] + p[i2]);
#pragma unroll
    for (int e = 0; e < EE; ++e) comb[(size_t)n * EE + e] = 0.0f;
    comb[(size_t)n * EE + i1] = p[i1] * ws;
    comb[(size_t)n * EE + i2] = p[i2] * ws;
#pragma unroll
    for (int e = 0; e < EE; ++e) atomicAdd(&auxP[e], p[e]);
    atomicAdd(&auxC[i1], 1.0f);
    atomicAdd(&auxC[i2], 1.0f);
}

__global__ void k_aux_final(const float* __restrict__ auxP, const float* __restrict__ auxC,
                            float* __restrict__ auxAcc) {
    if (threadIdx.x == 0 && blockIdx.x == 0) {
        float s = 0.f;
        for (int e = 0; e < EE; ++e) {
            float f = auxC[e] * (1.0f / ((float)NTOK * KTOP));
            float P = auxP[e] * (1.0f / (float)NTOK);
            s += f * P;
        }
        auxAcc[0] += (float)EE * s;
    }
}

// Mean-pool over S, LayerNorm, 128->2 head. One thread per batch element.
__global__ void __launch_bounds__(128)
k_head(const float* __restrict__ h, const float* __restrict__ g,
       const float* __restrict__ bta, const float* __restrict__ hw,
       const float* __restrict__ hb, const float* __restrict__ auxAcc,
       float* __restrict__ out) {
    int b = threadIdx.x;
    float pool[DD];
    for (int d = 0; d < DD; ++d) {
        float s = 0.f;
        for (int s2 = 0; s2 < SS; ++s2) s += h[((size_t)b * SS + s2) * DD + d];
        pool[d] = s * (1.0f / SS);
    }
    float m = 0.f;
    for (int d = 0; d < DD; ++d) m += pool[d];
    m *= (1.0f / DD);
    float v = 0.f;
    for (int d = 0; d < DD; ++d) { float t = pool[d] - m; v += t * t; }
    v *= (1.0f / DD);
    float rs = rsqrtf(v + 1e-5f);
    float o0 = hb[0], o1 = hb[1];
    for (int d = 0; d < DD; ++d) {
        float t = (pool[d] - m) * rs * g[d] + bta[d];
        o0 += t * hw[d * 2 + 0];
        o1 += t * hw[d * 2 + 1];
    }
    out[b * 2 + 0] = o0;
    out[b * 2 + 1] = o1;
    if (b == 0) out[2 * BB] = auxAcc[0];
}

// ---------------------------------------------------------------------------
// Host side
// ---------------------------------------------------------------------------
extern "C" void kernel_launch(void* const* d_in, const int* in_sizes, int n_in,
                              void* d_out, int out_size, void* d_ws, size_t ws_size,
                              hipStream_t stream) {
    (void)in_sizes; (void)n_in; (void)out_size; (void)ws_size;

    const float* x      = (const float*)d_in[0];
    const float* conv_w = (const float*)d_in[1];
    const float* bn_g   = (const float*)d_in[2];
    const float* bn_b   = (const float*)d_in[3];
    const float* pos    = (const float*)d_in[4];
    const float* ln1_g  = (const float*)d_in[5];
    const float* ln1_b  = (const float*)d_in[6];
    const float* wq     = (const float*)d_in[7];
    const float* wk     = (const float*)d_in[8];
    const float* wv     = (const float*)d_in[9];
    const float* wo     = (const float*)d_in[10];
    const float* bq     = (const float*)d_in[11];
    const float* bk     = (const float*)d_in[12];
    const float* bv     = (const float*)d_in[13];
    const float* bo     = (const float*)d_in[14];
    const float* ln2_g  = (const float*)d_in[15];
    const float* ln2_b  = (const float*)d_in[16];
    const float* gate_w = (const float*)d_in[17];
    const float* gate_b = (const float*)d_in[18];
    const float* w1     = (const float*)d_in[19];
    const float* b1     = (const float*)d_in[20];
    const float* w2     = (const float*)d_in[21];
    const float* b2     = (const float*)d_in[22];
    const float* hln_g  = (const float*)d_in[23];
    const float* hln_b  = (const float*)d_in[24];
    const float* head_w = (const float*)d_in[25];
    const float* head_b = (const float*)d_in[26];
    float* out = (float*)d_out;

    // ---- carve workspace ----
    char* base = (char*)d_ws;
    size_t off = 0;
    auto carve = [&](size_t bytes) -> char* {
        char* p = base + off;
        off += (bytes + 255) & ~(size_t)255;
        return p;
    };
    __bf16* xb    = (__bf16*)carve((size_t)BB * CIN * TT * 2);     // 57.7 MB
    __bf16* wcv   = (__bf16*)carve((size_t)CIN * 32 * DD * 2);     // 3.6 MB
    float*  hconv = (float*) carve((size_t)NTOK * DD * 4);         // 4 MB
    float*  bnst  = (float*) carve(2 * DD * 4);
    float*  h     = (float*) carve((size_t)NTOK * DD * 4);         // 4 MB
    __bf16* y     = (__bf16*)carve((size_t)NTOK * DD * 2);         // 2 MB
    float*  qbuf  = (float*) carve((size_t)NTOK * DD * 4);
    float*  kbuf  = (float*) carve((size_t)NTOK * DD * 4);
    float*  vbuf  = (float*) carve((size_t)NTOK * DD * 4);
    __bf16* att   = (__bf16*)carve((size_t)NTOK * DD * 2);
    __bf16* wqb   = (__bf16*)carve((size_t)LL * DD * DD * 2);
    __bf16* wkb   = (__bf16*)carve((size_t)LL * DD * DD * 2);
    __bf16* wvb   = (__bf16*)carve((size_t)LL * DD * DD * 2);
    __bf16* wob   = (__bf16*)carve((size_t)LL * DD * DD * 2);
    __bf16* w1b   = (__bf16*)carve((size_t)LL * EE * DD * HID * 2); // 4 MB
    __bf16* w2b   = (__bf16*)carve((size_t)LL * EE * HID * DD * 2); // 4 MB
    __bf16* hid   = (__bf16*)carve((size_t)NTOK * HID * 2);         // 8 MB
    float*  comb  = (float*) carve((size_t)NTOK * EE * 4);
    float*  aux   = (float*) carve(32 * 4);  // [0..7]=P, [8..15]=C, [16]=acc

    const int TPB = 256;
    auto blocks = [](size_t n, int tpb) { return (unsigned)((n + tpb - 1) / tpb); };

    // ---- prep: bf16 conversions + layouts ----
    k_prep_x<<<blocks((size_t)BB * CIN * TT, TPB), TPB, 0, stream>>>(x, xb);
    k_prep_convw<<<blocks((size_t)CIN * 32 * DD, TPB), TPB, 0, stream>>>(conv_w, wcv);
    k_cvt_f32_bf16<<<blocks(LL * DD * DD, TPB), TPB, 0, stream>>>(wq, wqb, LL * DD * DD);
    k_cvt_f32_bf16<<<blocks(LL * DD * DD, TPB), TPB, 0, stream>>>(wk, wkb, LL * DD * DD);
    k_cvt_f32_bf16<<<blocks(LL * DD * DD, TPB), TPB, 0, stream>>>(wv, wvb, LL * DD * DD);
    k_cvt_f32_bf16<<<blocks(LL * DD * DD, TPB), TPB, 0, stream>>>(wo, wob, LL * DD * DD);
    k_cvt_f32_bf16<<<blocks((size_t)LL * EE * DD * HID, TPB), TPB, 0, stream>>>(w1, w1b, LL * EE * DD * HID);
    k_cvt_f32_bf16<<<blocks((size_t)LL * EE * HID * DD, TPB), TPB, 0, stream>>>(w2, w2b, LL * EE * HID * DD);
    k_zero<<<1, 32, 0, stream>>>(aux, 32);

    // ---- conv frontend (implicit GEMM, WMMA) + BN + GELU + pos ----
    k_conv_gemm<<<dim3(DD / 64, BB), 128, 0, stream>>>(xb, wcv, hconv);
    k_bn_stats<<<DD, 256, 0, stream>>>(hconv, bnst);
    k_bn_act<<<blocks(NTOK * DD, TPB), TPB, 0, stream>>>(hconv, bnst, bn_g, bn_b, pos, h);

    // ---- transformer layers ----
    dim3 gTok(DD / 64, NTOK / 64);    // (2,128) for N=128 GEMMs
    dim3 gHid(HID / 64, NTOK / 64);   // (8,128) for N=512 GEMM
    for (int l = 0; l < LL; ++l) {
        // pre-LN attention
        k_ln<<<blocks(NTOK, TPB), TPB, 0, stream>>>(h, ln1_g + l * DD, ln1_b + l * DD, y);
        k_gemm<0><<<gTok, 128, 0, stream>>>(y, wqb + (size_t)l * DD * DD, bq + l * DD,
                                            qbuf, nullptr, nullptr, 0, NTOK, DD, DD);
        k_gemm<0><<<gTok, 128, 0, stream>>>(y, wkb + (size_t)l * DD * DD, bk + l * DD,
                                            kbuf, nullptr, nullptr, 0, NTOK, DD, DD);
        k_gemm<0><<<gTok, 128, 0, stream>>>(y, wvb + (size_t)l * DD * DD, bv + l * DD,
                                            vbuf, nullptr, nullptr, 0, NTOK, DD, DD);
        k_attn<<<BB * HEADS, 64, 0, stream>>>(qbuf, kbuf, vbuf, att);
        k_gemm<1><<<gTok, 128, 0, stream>>>(att, wob + (size_t)l * DD * DD, bo + l * DD,
                                            h, nullptr, nullptr, 0, NTOK, DD, DD);
        // pre-LN MoE
        k_ln<<<blocks(NTOK, TPB), TPB, 0, stream>>>(h, ln2_g + l * DD, ln2_b + l * DD, y);
        k_zero<<<1, 32, 0, stream>>>(aux, 16);  // reset P/C (keep acc)
        k_gate<<<blocks(NTOK, TPB), TPB, 0, stream>>>(y, gate_w + (size_t)l * DD * EE,
                                                      gate_b + l * EE, comb, aux, aux + 8);
        k_aux_final<<<1, 32, 0, stream>>>(aux, aux + 8, aux + 16);
        for (int e = 0; e < EE; ++e) {
            const size_t we = (size_t)(l * EE + e);
            k_gemm<2><<<gHid, 128, 0, stream>>>(y, w1b + we * DD * HID, b1 + we * HID,
                                                nullptr, hid, nullptr, 0, NTOK, HID, DD);
            k_gemm<3><<<gTok, 128, 0, stream>>>(hid, w2b + we * HID * DD, b2 + we * DD,
                                                h, nullptr, comb + e, EE, NTOK, DD, HID);
        }
    }

    // ---- pool + LN + head + aux output ----
    k_head<<<1, 128, 0, stream>>>(h, hln_g, hln_b, head_w, head_b, aux + 16, out);
}
